// RNN_84284438216716
// MI455X (gfx1250) — matrix operand order repaired and checked by reading, not compiled, not run on hardware
//
#include <hip/hip_runtime.h>
#include <math.h>

// Problem constants (fixed by the reference)
#define BATCH 64
#define TLEN  1024
#define DX    64
#define DZ    1024
#define NSEG  16      // TLEN / 64 reset segments
#define NSTEP 64      // steps per segment
#define ROWS  1024    // BATCH * NSEG folded rows

typedef __attribute__((ext_vector_type(16))) _Float16 v16h;
typedef __attribute__((ext_vector_type(8)))  _Float16 v8h;
typedef __attribute__((ext_vector_type(8)))  float    v8f;

union AFrag { v16h v; v8h h[2]; };

// Branch-free tanh on the native transcendental unit:
//   tanh(x) = sign(x) * (1 - 2 / (exp2(2*log2(e)*|x|) + 1))
// v_exp_f32 + v_rcp_f32, no EXEC manipulation; |x| large -> exp2 = +inf
// -> rcp = 0 -> result saturates to +-1 exactly.
__device__ __forceinline__ float fast_tanh(float x) {
    const float ax = __builtin_fabsf(x);
    const float e  = __builtin_amdgcn_exp2f(ax * 2.8853900817779268f); // 2*log2(e)
    const float r  = 1.0f - 2.0f * __builtin_amdgcn_rcpf(e + 1.0f);
    return __builtin_copysignf(r, x);
}

// ---------------------------------------------------------------------------
// K1: Gram = Bw @ Bw^T (64x64), then Gauss-Jordan invert in LDS -> Ginv
// ---------------------------------------------------------------------------
__global__ __launch_bounds__(256)
void gram_inv_kernel(const float* __restrict__ Bw, float* __restrict__ Ginv) {
    __shared__ float A[64][129];   // augmented [G | I]
    __shared__ float fac[64];
    const int tid = threadIdx.x;

    for (int e = tid; e < 64 * 64; e += 256) {
        const int i = e >> 6, j = e & 63;
        float s = 0.f;
        const float* ri = Bw + i * DZ;
        const float* rj = Bw + j * DZ;
        for (int k = 0; k < DZ; ++k) s += ri[k] * rj[k];
        A[i][j]      = s;
        A[i][64 + j] = (i == j) ? 1.f : 0.f;
    }
    __syncthreads();

    for (int col = 0; col < 64; ++col) {
        const float inv = 1.0f / A[col][col];
        __syncthreads();
        if (tid < 128) A[col][tid] *= inv;
        __syncthreads();
        if (tid < 64) fac[tid] = A[tid][col];
        __syncthreads();
        for (int e = tid; e < 64 * 128; e += 256) {
            const int r = e >> 7, c = e & 127;
            if (r != col) A[r][c] -= fac[r] * A[col][c];
        }
        __syncthreads();
    }
    for (int e = tid; e < 64 * 64; e += 256)
        Ginv[e] = A[e >> 6][64 + (e & 63)];
}

// ---------------------------------------------------------------------------
// K2: Wt[n][k] = (f16) W_hh[n][k]   (n-major so WMMA B-frags are contiguous in k)
// ---------------------------------------------------------------------------
__global__ __launch_bounds__(256)
void prep_w_kernel(const float* __restrict__ W, _Float16* __restrict__ Wt) {
    const int idx = blockIdx.x * 256 + threadIdx.x;   // 1M elements
    const int k = idx >> 10, n = idx & 1023;
    Wt[n * DZ + k] = (_Float16)W[n * DZ + k];
}

// ---------------------------------------------------------------------------
// K3: reset states, only at t % 64 == 0:
//     zr[p][:] = Bw^T @ (Ginv @ x[b, s*64, :]),  p = b*16 + s
// ---------------------------------------------------------------------------
__global__ __launch_bounds__(256)
void reset_kernel(const float* __restrict__ x, const float* __restrict__ Bw,
                  const float* __restrict__ Ginv, float* __restrict__ zr) {
    const int p = blockIdx.x;                 // 0..1023
    const int b = p >> 4, s = p & 15;
    const float* xv = x + ((size_t)(b * TLEN + s * NSTEP)) * DX;
    __shared__ float xs[64];
    __shared__ float u[64];
    const int tid = threadIdx.x;
    if (tid < 64) xs[tid] = xv[tid];
    __syncthreads();
    if (tid < 64) {
        float a = 0.f;
        for (int i = 0; i < 64; ++i) a += Ginv[tid * 64 + i] * xs[i];
        u[tid] = a;
    }
    __syncthreads();
    for (int z = tid; z < DZ; z += 256) {
        float a = 0.f;
        for (int i = 0; i < 64; ++i) a += Bw[i * DZ + z] * u[i];
        zr[(size_t)p * DZ + z] = a;
    }
}

// ---------------------------------------------------------------------------
// K4: one RNN step:  out[:, t=k] = tanh(Hprev @ W_hh^T + bias)
//     Hprev rows live in zr (k==0) or out at t=k-1.  1024x1024x1024 GEMM.
//     64 blocks, 128x128 tile, 8 waves (2 M x 4 N), each wave 4x2 WMMA tiles.
// ---------------------------------------------------------------------------
__global__ __launch_bounds__(256)
void rnn_step_kernel(const float* __restrict__ zr, const _Float16* __restrict__ Wt,
                     const float* __restrict__ b_ih, const float* __restrict__ b_hh,
                     float* out, int k) {
    const int m0 = (blockIdx.x >> 3) * 128;
    const int n0 = (blockIdx.x & 7) * 128;
    const int tid  = threadIdx.x;
    const int wave = tid >> 5;
    const int lane = tid & 31;
    const int wm = wave & 1;     // 0..1 -> 64-row slab
    const int wn = wave >> 1;    // 0..3 -> 32-col slab

    __shared__ _Float16 sA[128][40];   // 40-half pitch: conflict-free b128 reads
    __shared__ float    sBias[128];

    if (tid < 128) sBias[tid] = b_ih[n0 + tid] + b_hh[n0 + tid];

    v8f acc[4][2] = {};

    for (int kk = 0; kk < DZ; kk += 32) {
        // ---- stage A tile (128 rows x 32 K) fp32 -> f16 into LDS ----
        {
            const int r  = tid >> 1;            // 0..127
            const int cb = (tid & 1) * 16;      // half-row of 16
            const int p  = m0 + r;              // folded row: b*16 + s
            const float* src;
            if (k == 0) {
                src = zr + (size_t)p * DZ;
            } else {
                const int b = p >> 4, s = p & 15;
                src = out + ((size_t)(b * TLEN + s * NSTEP + (k - 1))) * DZ;
            }
            const float4* s4 = (const float4*)(src + kk + cb);
#pragma unroll
            for (int q = 0; q < 4; ++q) {
                float4 v = s4[q];
                sA[r][cb + q * 4 + 0] = (_Float16)v.x;
                sA[r][cb + q * 4 + 1] = (_Float16)v.y;
                sA[r][cb + q * 4 + 2] = (_Float16)v.z;
                sA[r][cb + q * 4 + 3] = (_Float16)v.w;
            }
        }
        __syncthreads();

        // ---- B fragments straight from L2-resident Wt: lane&15 = N col ----
        const _Float16* bp = Wt + (size_t)(n0 + wn * 32 + (lane & 15)) * DZ + kk;
        const int boff = (lane >> 4) * 8;
        AFrag bf0, bf1;
        bf0.h[0] = *(const v8h*)(bp + boff);
        bf0.h[1] = *(const v8h*)(bp + 16 + boff);
        bf1.h[0] = *(const v8h*)(bp + 16 * DZ + boff);
        bf1.h[1] = *(const v8h*)(bp + 16 * DZ + 16 + boff);
        __builtin_prefetch(bp + 32, 0, 1);   // next K-chunk of W

        // ---- A fragments from LDS + 8 WMMAs ----
        const int off = (lane >> 4) * 8;   // lanes 0-15: K0-7/16-23, 16-31: K8-15/24-31
#pragma unroll
        for (int mt = 0; mt < 4; ++mt) {
            const int row = wm * 64 + mt * 16 + (lane & 15);
            AFrag af;
            af.h[0] = *(const v8h*)&sA[row][off];
            af.h[1] = *(const v8h*)&sA[row][16 + off];
            acc[mt][0] = __builtin_amdgcn_wmma_f32_16x16x32_f16(
                false, af.v, false, bf0.v, (short)0, acc[mt][0], false, false);
            acc[mt][1] = __builtin_amdgcn_wmma_f32_16x16x32_f16(
                false, af.v, false, bf1.v, (short)0, acc[mt][1], false, false);
        }
        __syncthreads();
    }

    // ---- epilogue: bias + branch-free tanh, scatter to out[:, t=k] ----
    // C/D layout: VGPR r, lanes 0-15 -> M=r, lanes 16-31 -> M=r+8; N = lane&15
    const int mbase = m0 + wm * 64;
    const int nloc  = wn * 32;
#pragma unroll
    for (int mt = 0; mt < 4; ++mt) {
#pragma unroll
        for (int nt = 0; nt < 2; ++nt) {
#pragma unroll
            for (int r = 0; r < 8; ++r) {
                const int mm = mbase + mt * 16 + r + ((lane >> 4) << 3);
                const int nn = nloc + nt * 16 + (lane & 15);
                const float v = fast_tanh(acc[mt][nt][r] + sBias[nn]);
                const int b = mm >> 4, s = mm & 15;
                out[((size_t)(b * TLEN + s * NSTEP + k)) * DZ + (n0 + nn)] = v;
            }
        }
    }
}

// ---------------------------------------------------------------------------
extern "C" void kernel_launch(void* const* d_in, const int* in_sizes, int n_in,
                              void* d_out, int out_size, void* d_ws, size_t ws_size,
                              hipStream_t stream) {
    (void)in_sizes; (void)n_in; (void)out_size; (void)ws_size;
    const float* x    = (const float*)d_in[0];   // (64,1024,64)
    const float* Bw   = (const float*)d_in[1];   // (64,1024)
    const float* W_hh = (const float*)d_in[2];   // (1024,1024)
    const float* b_ih = (const float*)d_in[3];   // (1024)
    const float* b_hh = (const float*)d_in[4];   // (1024)
    // d_in[5] = n (==64, hardcoded)
    float* out = (float*)d_out;

    char* ws = (char*)d_ws;
    float*    Ginv = (float*)ws;                                   // 16 KB
    _Float16* Wt   = (_Float16*)(ws + 16384);                      // 2 MB
    float*    zr   = (float*)(ws + 16384 + (size_t)DZ * DZ * 2);   // 4 MB

    gram_inv_kernel<<<1, 256, 0, stream>>>(Bw, Ginv);
    prep_w_kernel<<<(DZ * DZ) / 256, 256, 0, stream>>>(W_hh, Wt);
    reset_kernel<<<ROWS, 256, 0, stream>>>(x, Bw, Ginv, zr);

    for (int k = 0; k < NSTEP; ++k)
        rnn_step_kernel<<<64, 256, 0, stream>>>(zr, Wt, b_ih, b_hh, out, k);
}
